// SimpleGPT_88725434400974
// MI455X (gfx1250) — matrix operand order repaired and checked by reading, not compiled, not run on hardware
//
#include <hip/hip_runtime.h>
#include <math.h>

// ---------------------------------------------------------------------------
// SimpleGPT forward for MI455X (gfx1250): bf16 WMMA everywhere, f32 accum.
// ---------------------------------------------------------------------------

typedef __attribute__((ext_vector_type(16))) __bf16 v16bf;
typedef __attribute__((ext_vector_type(8)))  __bf16 bf16x8;
typedef __attribute__((ext_vector_type(8)))  float  v8f;

#define B_SZ   64
#define T_SZ   50
#define V_SZ   32000
#define E_SZ   64
#define DH_SZ  128
#define H_SZ   8
#define FF_SZ  256
#define BT_SZ  (B_SZ * T_SZ)      // 3200 rows
#define HD_SZ  (H_SZ * DH_SZ)     // 1024

static __device__ __forceinline__ float bf2f(__bf16 x) {
  unsigned int v = ((unsigned int)__builtin_bit_cast(unsigned short, x)) << 16;
  return __builtin_bit_cast(float, v);
}
static __device__ __forceinline__ __bf16 f2bf(float f) {
  unsigned int u = __builtin_bit_cast(unsigned int, f);
  u += 0x7FFFu + ((u >> 16) & 1u);              // round-to-nearest-even
  return __builtin_bit_cast(__bf16, (unsigned short)(u >> 16));
}

// ---------------------------------------------------------------------------
// 1) Embedding: h[bt, e] = tok_emb[x[bt], e] + pos_emb[t, e]  ->  bf16
// ---------------------------------------------------------------------------
__global__ void embed_k(const int* __restrict__ x, const float* __restrict__ tok,
                        const float* __restrict__ pos, __bf16* __restrict__ h) {
  int idx = blockIdx.x * blockDim.x + threadIdx.x;
  if (idx >= BT_SZ * E_SZ) return;
  int bt = idx / E_SZ;
  int e  = idx - bt * E_SZ;
  int t  = bt % T_SZ;
  h[idx] = f2bf(tok[(size_t)x[bt] * E_SZ + e] + pos[t * E_SZ + e]);
}

// ---------------------------------------------------------------------------
// 2) Weight pack: f32 source -> bf16 WMMA B-fragments (32x16 per K-step).
//    Fragment layout used here, contiguous k (16-bit B 32x16): lane n%16 = column,
//    lanes 0-15 hold K=0..15, lanes 16-31 hold K=16..31, element e -> K=kbase+e.
//    Stored lane-major: dst[(frag*32 + lane)*16 + e], frag = nt*KT + kt.
//    src index = (n / nPerHead)*headStride + k*kStride + (n % nPerHead)
//      plain [K,N]:  kStride=N,  nPerHead=N,  headStride=0
//      Wq [H,E,DH]:  kStride=DH, nPerHead=DH, headStride=E*DH
// ---------------------------------------------------------------------------
__global__ void pack_w(const float* __restrict__ src, __bf16* __restrict__ dst,
                       int K, int N, int kStride, int nPerHead, int headStride) {
  int id = blockIdx.x * blockDim.x + threadIdx.x;
  int KT = K >> 5, NT = N >> 4;
  if (id >= KT * NT * 32) return;
  int lane = id & 31;
  int frag = id >> 5;
  int kt = frag % KT;
  int nt = frag / KT;
  int n  = nt * 16 + (lane & 15);
  int kb = kt * 32 + ((lane >> 4) << 4);
  int hidx = n / nPerHead;
  int nin  = n % nPerHead;
  __bf16* d = dst + (size_t)id * 16;
#pragma unroll
  for (int e = 0; e < 16; ++e)
    d[e] = f2bf(src[(size_t)hidx * headStride + (size_t)(kb + e) * kStride + nin]);
}

// ---------------------------------------------------------------------------
// 3) GEMM: C[M,N] = A[M,K] @ B[K,N] + bias (+ optional exact GELU).
//    A: row-major bf16. B: pre-packed fragments. 8 waves/block, each wave
//    computes a 16x64 tile (1 A-frag reused by 4 WMMAs per 32-wide k-step).
//    Requires M%16==0, N%64==0, K%32==0 (true for every call site).
// ---------------------------------------------------------------------------
__global__ void __launch_bounds__(256)
gemm_bf16_wmma(const __bf16* __restrict__ A, const __bf16* __restrict__ Bp,
               const float* __restrict__ bias,
               float* __restrict__ outF, __bf16* __restrict__ outB,
               int M, int N, int K, int act_gelu) {
  const int lane  = threadIdx.x & 31;
  const int wave  = threadIdx.x >> 5;
  const int KT    = K >> 5;
  const int mtile = blockIdx.x * 8 + wave;
  if (mtile * 16 >= M) return;                 // wave-uniform: EXEC stays full
  const int nt0 = blockIdx.y * 4;

  v8f acc[4] = {};
  const int mrow  = mtile * 16 + (lane & 15);
  const int kbase = (lane >> 4) << 4;          // A-frag: hi lanes own K+16 (contiguous)
  const __bf16* arow = A + (size_t)mrow * K;

  for (int kt = 0; kt < KT; ++kt) {
    const __bf16* ap = arow + (kt << 5);
    bf16x8 a0 = *(const bf16x8*)(ap + kbase);        // K = kbase..kbase+7
    bf16x8 a1 = *(const bf16x8*)(ap + 8 + kbase);    // K = kbase+8..kbase+15
    v16bf af;
#pragma unroll
    for (int i = 0; i < 8; ++i) { af[i] = a0[i]; af[8 + i] = a1[i]; }
#pragma unroll
    for (int j = 0; j < 4; ++j) {
      const __bf16* bp = Bp + (((size_t)(nt0 + j) * KT + kt) * 32 + lane) * 16;
      bf16x8 b0 = *(const bf16x8*)bp;
      bf16x8 b1 = *(const bf16x8*)(bp + 8);
      v16bf bfr;
#pragma unroll
      for (int i = 0; i < 8; ++i) { bfr[i] = b0[i]; bfr[8 + i] = b1[i]; }
      acc[j] = __builtin_amdgcn_wmma_f32_16x16x32_bf16(
          false, af, false, bfr, (short)0, acc[j], false, false);
    }
  }

  const int lm = (lane >> 4) << 3;             // C layout: hi lanes own M+8
  const int ln = lane & 15;
#pragma unroll
  for (int j = 0; j < 4; ++j) {
    const int ncol = (nt0 + j) * 16 + ln;
    const float bv = bias ? bias[ncol] : 0.0f;
#pragma unroll
    for (int r = 0; r < 8; ++r) {
      const int row = mtile * 16 + lm + r;
      float v = acc[j][r] + bv;
      if (act_gelu) v = 0.5f * v * (1.0f + erff(v * 0.70710678118654752f));
      if (outF) outF[(size_t)row * N + ncol] = v;
      if (outB) outB[(size_t)row * N + ncol] = f2bf(v);
    }
  }
}

// ---------------------------------------------------------------------------
// 4) Attention: one 64-thread block per (b, h, t). Scores + causal mask +
//    LDS softmax + AV. Output written head-concat [BT, H*DH] in bf16.
// ---------------------------------------------------------------------------
__global__ void __launch_bounds__(64)
attn_k(const __bf16* __restrict__ Q, const __bf16* __restrict__ Km,
       const __bf16* __restrict__ Vm, __bf16* __restrict__ O) {
  __shared__ float sc[64];
  __shared__ float red[64];
  const int s   = threadIdx.x;
  const int bid = blockIdx.x;                  // (b*H + h)*T + t
  const int t   = bid % T_SZ;
  const int bh  = bid / T_SZ;
  const int hh  = bh % H_SZ;
  const int b   = bh / H_SZ;

  const size_t rowbase = (size_t)(b * T_SZ) * HD_SZ + (size_t)hh * DH_SZ;
  const size_t qoff    = rowbase + (size_t)t * HD_SZ;

  float sacc = 0.0f;
  if (s < T_SZ) {
    const bf16x8* qp = (const bf16x8*)(Q + qoff);
    const bf16x8* kp = (const bf16x8*)(Km + rowbase + (size_t)s * HD_SZ);
#pragma unroll
    for (int c = 0; c < DH_SZ / 8; ++c) {
      bf16x8 qv = qp[c], kv = kp[c];
#pragma unroll
      for (int i = 0; i < 8; ++i) sacc += bf2f(qv[i]) * bf2f(kv[i]);
    }
    sacc *= 0.08838834764831845f;              // 1/sqrt(128)
  }
  float score = (s <= t && s < T_SZ) ? sacc : -__builtin_inff();
  sc[s] = score; red[s] = score;
  __syncthreads();
  for (int st = 32; st > 0; st >>= 1) {
    if (s < st) red[s] = fmaxf(red[s], red[s + st]);
    __syncthreads();
  }
  const float mx = red[0];
  __syncthreads();
  const float ex = __expf(sc[s] - mx);         // masked lanes -> 0
  red[s] = ex;
  __syncthreads();
  for (int st = 32; st > 0; st >>= 1) {
    if (s < st) red[s] += red[s + st];
    __syncthreads();
  }
  const float inv = 1.0f / red[0];
  __syncthreads();
  sc[s] = ex * inv;
  __syncthreads();

  float a0 = 0.0f, a1 = 0.0f;
  const __bf16* vb = Vm + rowbase;
  for (int ss = 0; ss < T_SZ; ++ss) {
    const float w = sc[ss];
    a0 += w * bf2f(vb[(size_t)ss * HD_SZ + s]);
    a1 += w * bf2f(vb[(size_t)ss * HD_SZ + s + 64]);
  }
  __bf16* op = O + qoff;
  op[s]      = f2bf(a0);
  op[s + 64] = f2bf(a1);
}

// ---------------------------------------------------------------------------
// Launch
// ---------------------------------------------------------------------------
extern "C" void kernel_launch(void* const* d_in, const int* in_sizes, int n_in,
                              void* d_out, int out_size, void* d_ws, size_t ws_size,
                              hipStream_t stream) {
  (void)in_sizes; (void)n_in; (void)out_size; (void)ws_size;
  const int*   x   = (const int*)  d_in[0];
  const float* tok = (const float*)d_in[1];
  const float* pos = (const float*)d_in[2];
  const float* Wq  = (const float*)d_in[3];
  const float* bq  = (const float*)d_in[4];
  const float* Wk  = (const float*)d_in[5];
  const float* bk  = (const float*)d_in[6];
  const float* Wv  = (const float*)d_in[7];
  const float* bv  = (const float*)d_in[8];
  const float* Wo  = (const float*)d_in[9];
  const float* bo  = (const float*)d_in[10];
  const float* W1  = (const float*)d_in[11];
  const float* b1  = (const float*)d_in[12];
  const float* W2  = (const float*)d_in[13];
  const float* b2  = (const float*)d_in[14];
  const float* Wf  = (const float*)d_in[15];
  const float* bfb = (const float*)d_in[16];
  float* logits = (float*)d_out;

  char* ws = (char*)d_ws;
  size_t off = 0;
  auto take = [&](size_t bytes) -> char* {
    char* p = ws + off;
    off = (off + bytes + 255) & ~(size_t)255;
    return p;
  };

  __bf16* hbf = (__bf16*)take((size_t)BT_SZ * E_SZ * 2);
  __bf16* pWq = (__bf16*)take((size_t)E_SZ * HD_SZ * 2);
  __bf16* pWk = (__bf16*)take((size_t)E_SZ * HD_SZ * 2);
  __bf16* pWv = (__bf16*)take((size_t)E_SZ * HD_SZ * 2);
  __bf16* Qb  = (__bf16*)take((size_t)BT_SZ * HD_SZ * 2);
  __bf16* Kb  = (__bf16*)take((size_t)BT_SZ * HD_SZ * 2);
  __bf16* Vb  = (__bf16*)take((size_t)BT_SZ * HD_SZ * 2);
  __bf16* Ob  = (__bf16*)take((size_t)BT_SZ * HD_SZ * 2);
  __bf16* pWo = (__bf16*)take((size_t)HD_SZ * E_SZ * 2);
  __bf16* Ab  = (__bf16*)take((size_t)BT_SZ * E_SZ * 2);
  __bf16* pW1 = (__bf16*)take((size_t)E_SZ * FF_SZ * 2);
  __bf16* Mb  = (__bf16*)take((size_t)BT_SZ * FF_SZ * 2);
  __bf16* pW2 = (__bf16*)take((size_t)FF_SZ * E_SZ * 2);
  __bf16* Yb  = (__bf16*)take((size_t)BT_SZ * E_SZ * 2);
  __bf16* pWf = (__bf16*)take((size_t)E_SZ * V_SZ * 2);

  // 1) embeddings
  embed_k<<<(BT_SZ * E_SZ + 255) / 256, 256, 0, stream>>>(x, tok, pos, hbf);

  // 2) weight packing (all independent of activations)
  auto packBlocks = [](int K, int N) { return ((K >> 5) * (N >> 4) * 32 + 255) / 256; };
  pack_w<<<packBlocks(E_SZ, HD_SZ), 256, 0, stream>>>(Wq, pWq, E_SZ, HD_SZ, DH_SZ, DH_SZ, E_SZ * DH_SZ);
  pack_w<<<packBlocks(E_SZ, HD_SZ), 256, 0, stream>>>(Wk, pWk, E_SZ, HD_SZ, DH_SZ, DH_SZ, E_SZ * DH_SZ);
  pack_w<<<packBlocks(E_SZ, HD_SZ), 256, 0, stream>>>(Wv, pWv, E_SZ, HD_SZ, DH_SZ, DH_SZ, E_SZ * DH_SZ);
  pack_w<<<packBlocks(HD_SZ, E_SZ), 256, 0, stream>>>(Wo, pWo, HD_SZ, E_SZ, E_SZ, E_SZ, 0);
  pack_w<<<packBlocks(E_SZ, FF_SZ), 256, 0, stream>>>(W1, pW1, E_SZ, FF_SZ, FF_SZ, FF_SZ, 0);
  pack_w<<<packBlocks(FF_SZ, E_SZ), 256, 0, stream>>>(W2, pW2, FF_SZ, E_SZ, E_SZ, E_SZ, 0);
  pack_w<<<packBlocks(E_SZ, V_SZ), 256, 0, stream>>>(Wf, pWf, E_SZ, V_SZ, V_SZ, V_SZ, 0);

  const dim3 blk(256);
  const int MB = BT_SZ / 128;                  // 25 blocks of 8 waves x 16 rows

  // 3) QKV projections (fused over heads: N = H*DH = 1024)
  gemm_bf16_wmma<<<dim3(MB, HD_SZ / 64), blk, 0, stream>>>(hbf, pWq, bq, nullptr, Qb, BT_SZ, HD_SZ, E_SZ, 0);
  gemm_bf16_wmma<<<dim3(MB, HD_SZ / 64), blk, 0, stream>>>(hbf, pWk, bk, nullptr, Kb, BT_SZ, HD_SZ, E_SZ, 0);
  gemm_bf16_wmma<<<dim3(MB, HD_SZ / 64), blk, 0, stream>>>(hbf, pWv, bv, nullptr, Vb, BT_SZ, HD_SZ, E_SZ, 0);

  // 4) attention -> head-concat output
  attn_k<<<B_SZ * H_SZ * T_SZ, 64, 0, stream>>>(Qb, Kb, Vb, Ob);

  // 5) output proj, FFN, vocab projection
  gemm_bf16_wmma<<<dim3(MB, E_SZ / 64),  blk, 0, stream>>>(Ob, pWo, bo, nullptr, Ab, BT_SZ, E_SZ, HD_SZ, 0);
  gemm_bf16_wmma<<<dim3(MB, FF_SZ / 64), blk, 0, stream>>>(Ab, pW1, b1, nullptr, Mb, BT_SZ, FF_SZ, E_SZ, 1);
  gemm_bf16_wmma<<<dim3(MB, E_SZ / 64),  blk, 0, stream>>>(Mb, pW2, b2, nullptr, Yb, BT_SZ, E_SZ, FF_SZ, 0);
  gemm_bf16_wmma<<<dim3(MB, V_SZ / 64),  blk, 0, stream>>>(Yb, pWf, bfb, logits, nullptr, BT_SZ, V_SZ, E_SZ, 0);
}
